// FlatLstm_18657337934502
// MI455X (gfx1250) — matrix operand order repaired and checked by reading, not compiled, not run on hardware
//
#include <hip/hip_runtime.h>

// Problem sizes (from the reference)
#define NN 4096
#define SS 8
#define II 256
#define LL 512
#define MM 256
#define DD (II + LL)   // 768

#define KSTEPS (DD / 32)      // 24
#define LDS_ROW_US 40         // 80 bytes per 32-k bf16 row (64B data + 16B pad)

typedef __attribute__((ext_vector_type(8)))  __bf16 v8bf;
typedef __attribute__((ext_vector_type(16))) __bf16 v16bf;
typedef __attribute__((ext_vector_type(8)))  float  v8f;

union FragBF16 { v16bf f; v8bf h[2]; };

__device__ __forceinline__ unsigned short f32_to_bf16(float x) {
  unsigned int u = __float_as_uint(x);
  u += 0x7FFFu + ((u >> 16) & 1u);      // round-to-nearest-even
  return (unsigned short)(u >> 16);
}

__device__ __forceinline__ float fast_sigmoid(float x) {
  return 1.0f / (1.0f + __expf(-x));
}
__device__ __forceinline__ float fast_tanh(float x) {
  return 2.0f / (1.0f + __expf(-2.0f * x)) - 1.0f;
}

// ---------------------------------------------------------------------------
// Pass 0a: build bf16 xh = concat([x_s, h_s]) per (n,s) row, fused with cvt.
// grid = (1, N*S), block = 192; 4 elements (16B load / 8B store) per thread.
// ---------------------------------------------------------------------------
__global__ void build_xh_bf16(const float* __restrict__ mod,
                              const float* __restrict__ h0,
                              unsigned short* __restrict__ xh) {
  const int r  = blockIdx.y;                 // [0, N*S)
  const int d0 = threadIdx.x * 4;            // [0, 768), multiple of 4
  const int n = r >> 3, s = r & 7;
  float4 v;
  if (d0 < II) v = *(const float4*)(mod + (size_t)n * (SS * II) + s * II + d0);
  else         v = *(const float4*)(h0  + (size_t)n * (SS * LL) + s * LL + (d0 - II));
  ushort4 o;
  o.x = f32_to_bf16(v.x); o.y = f32_to_bf16(v.y);
  o.z = f32_to_bf16(v.z); o.w = f32_to_bf16(v.w);
  *(ushort4*)(xh + (size_t)r * DD + d0) = o;
}

// ---------------------------------------------------------------------------
// Pass 0b: fp32 -> bf16 conversion, 4 elements per thread.
// ---------------------------------------------------------------------------
__global__ void conv_bf16x4(const float* __restrict__ src,
                            unsigned short* __restrict__ dst, int count) {
  int i0 = (blockIdx.x * blockDim.x + threadIdx.x) * 4;
  if (i0 < count) {
    float4 v = *(const float4*)(src + i0);
    ushort4 o;
    o.x = f32_to_bf16(v.x); o.y = f32_to_bf16(v.y);
    o.z = f32_to_bf16(v.z); o.w = f32_to_bf16(v.w);
    *(ushort4*)(dst + i0) = o;
  }
}

// ---------------------------------------------------------------------------
// Pass 1: 4-gate GEMM + LSTM cell. bf16 WMMA, f32 accumulate.
// grid = (L/64, N/128, S); block = 256 (8 wave32).
// B staged through double-buffered LDS via global_load_async_to_lds_b128.
// A fragments double-buffered across k-steps. B fragments prefetched at
// distance 2 through a 4-slot register ring; sched_group_barrier pins an
// interleaved DS-read / WMMA schedule so LDS latency is overlapped.
// ---------------------------------------------------------------------------
__global__ __launch_bounds__(256)
void lstm_gates_wmma(const unsigned short* __restrict__ xh,    // [N*S, D] bf16
                     const unsigned short* __restrict__ w4,    // [4,S,L,D] bf16
                     const float* __restrict__ bi, const float* __restrict__ bff,
                     const float* __restrict__ bg, const float* __restrict__ bo,
                     const float* __restrict__ c0,             // [N, S*L] f32
                     unsigned short* __restrict__ hnew) {      // [N, S*L] bf16
  __shared__ __attribute__((aligned(16))) unsigned short smem[2 * 256 * LDS_ROW_US];

  const int s      = blockIdx.z;
  const int l0     = blockIdx.x * 64;
  const int n0     = blockIdx.y * 128;
  const int tid    = threadIdx.x;
  const int wave   = tid >> 5;
  const int lane   = tid & 31;
  const int laneLo = lane & 15;
  const int hiHalf = lane >> 4;        // 0 or 1

  // ---- async B-stage writer: thread tid owns LDS row tid = gate*64 + lpos
  const int wg = tid >> 6, wl = tid & 63;
  const unsigned short* wsrc =
      w4 + (((size_t)wg * SS + s) * LL + (l0 + wl)) * DD;      // + k0 later
  const unsigned lds_w0 = (unsigned)(uintptr_t)&smem[(0 * 256 + tid) * LDS_ROW_US];
  const unsigned lds_w1 = (unsigned)(uintptr_t)&smem[(1 * 256 + tid) * LDS_ROW_US];

  // ---- A fragment addressing (16x32 bf16): lanes 0-15 -> K 0..15,
  // lanes 16-31 -> K 16..31; M = lane&15 within the wave tile.
  const int arow = n0 + wave * 16 + laneLo;
  const unsigned short* aptr = xh + ((size_t)arow * SS + s) * DD;
  const int ak = hiHalf ? 16 : 0;

  // ---- B fragment reader bases (per double-buffer half); row = slot*16+laneLo,
  // lanes 0-15 -> K 0..15, lanes 16-31 -> K 16..31 (32B = 16 ushort offset).
  const unsigned short* bbase0 = &smem[0] + (laneLo * LDS_ROW_US) + hiHalf * 16;
  const unsigned short* bbase1 = bbase0 + 256 * LDS_ROW_US;

  v8f acc[4][4];
  const v8f vzero = {0.f, 0.f, 0.f, 0.f, 0.f, 0.f, 0.f, 0.f};
  #pragma unroll
  for (int g = 0; g < 4; ++g)
    #pragma unroll
    for (int t = 0; t < 4; ++t) acc[g][t] = vzero;

  // One k-step: consume aCur + LDS buffer at bbase; prefetch next A tile and
  // issue the next async stage (into ldsNext) before the barrier.
  auto kstep = [&](int k0, FragBF16& aCur, FragBF16& aNx,
                   const unsigned short* bbase, bool notLast, unsigned ldsNext) {
    if (notLast) {
      aNx.h[0] = *(const v8bf*)(aptr + k0 + 32 + ak);
      aNx.h[1] = *(const v8bf*)(aptr + k0 + 32 + ak + 8);
      unsigned long long ga_ = (unsigned long long)(uintptr_t)(wsrc + k0 + 32);
      asm volatile(
          "global_load_async_to_lds_b128 %0, %1, off\n\t"
          "global_load_async_to_lds_b128 %0, %1, off offset:16\n\t"
          "global_load_async_to_lds_b128 %0, %1, off offset:32\n\t"
          "global_load_async_to_lds_b128 %0, %1, off offset:48"
          :: "v"(ldsNext), "v"(ga_) : "memory");
      asm volatile("s_wait_asynccnt 0x4" ::: "memory");  // current stage landed
    } else {
      asm volatile("s_wait_asynccnt 0x0" ::: "memory");
    }
    __syncthreads();   // publish current buffer to all waves

    // 4-slot register ring, prefetch distance 2 (3 fragments live at once ->
    // the allocator cannot collapse them into one buffer).
    FragBF16 b[4];
    b[0].h[0] = *(const v8bf*)(bbase);
    b[0].h[1] = *(const v8bf*)(bbase + 8);
    b[1].h[0] = *(const v8bf*)(bbase + (16 * LDS_ROW_US));
    b[1].h[1] = *(const v8bf*)(bbase + (16 * LDS_ROW_US) + 8);
    #pragma unroll
    for (int slot = 0; slot < 16; ++slot) {
      if (slot + 2 < 16) {
        const unsigned short* bp = bbase + (slot + 2) * (16 * LDS_ROW_US);
        b[(slot + 2) & 3].h[0] = *(const v8bf*)(bp);
        b[(slot + 2) & 3].h[1] = *(const v8bf*)(bp + 8);
      }
      acc[slot >> 2][slot & 3] = __builtin_amdgcn_wmma_f32_16x16x32_bf16(
          false, aCur.f, false, b[slot & 3].f, (short)0,
          acc[slot >> 2][slot & 3], false, false);
    }
#if __has_builtin(__builtin_amdgcn_sched_group_barrier)
    // Pin schedule in this region: 4 DS reads ahead, then 1 WMMA : 2 DS reads.
    __builtin_amdgcn_sched_group_barrier(0x100, 4, 0);   // DS read x4
    #pragma unroll
    for (int i = 0; i < 14; ++i) {
      __builtin_amdgcn_sched_group_barrier(0x008, 1, 0); // WMMA x1
      __builtin_amdgcn_sched_group_barrier(0x100, 2, 0); // DS read x2
    }
    __builtin_amdgcn_sched_group_barrier(0x008, 2, 0);   // final WMMA x2
#endif
    __syncthreads();   // all waves done reading before buffer is re-filled
  };

  // Prologue: stage k-step 0 into buffer 0, load first A fragment.
  {
    unsigned long long ga_ = (unsigned long long)(uintptr_t)wsrc;
    asm volatile(
        "global_load_async_to_lds_b128 %0, %1, off\n\t"
        "global_load_async_to_lds_b128 %0, %1, off offset:16\n\t"
        "global_load_async_to_lds_b128 %0, %1, off offset:32\n\t"
        "global_load_async_to_lds_b128 %0, %1, off offset:48"
        :: "v"(lds_w0), "v"(ga_) : "memory");
  }
  FragBF16 aA, aB;
  aA.h[0] = *(const v8bf*)(aptr + ak);
  aA.h[1] = *(const v8bf*)(aptr + ak + 8);

  for (int kk = 0; kk < KSTEPS; kk += 2) {
    kstep((kk + 0) * 32, aA, aB, bbase0, true,               lds_w1);
    kstep((kk + 1) * 32, aB, aA, bbase1, (kk + 2 < KSTEPS),  lds_w0);
  }

  // Epilogue: C layout -> VGPR j holds M = j (+8 for lanes>=16), N = lane&15.
  #pragma unroll
  for (int t = 0; t < 4; ++t) {
    const int col = l0 + t * 16 + laneLo;
    const float vbi = bi [s * LL + col];
    const float vbf = bff[s * LL + col];
    const float vbg = bg [s * LL + col];
    const float vbo = bo [s * LL + col];
    #pragma unroll
    for (int j = 0; j < 8; ++j) {
      const int row = n0 + wave * 16 + j + (hiHalf ? 8 : 0);
      const size_t off = ((size_t)row * SS + s) * LL + col;
      const float iv = fast_sigmoid(acc[0][t][j] + vbi);
      const float fv = fast_sigmoid(acc[1][t][j] + vbf);
      const float gv = fast_tanh   (acc[2][t][j] + vbg);
      const float ov = fast_sigmoid(acc[3][t][j] + vbo);
      const float cn = fv * c0[off] + iv * gv;
      const float hn = ov * fast_tanh(cn);
      hnew[off] = f32_to_bf16(hn);
    }
  }
}

// ---------------------------------------------------------------------------
// Pass 2: y = h_new @ Wy^T + by, bf16 WMMA, fp32 output (~8% of FLOPs;
// direct global loads, 2-deep B pipelining).
// grid = (M/64, N/128, S); block = 256 (8 wave32).
// ---------------------------------------------------------------------------
__global__ __launch_bounds__(256)
void lstm_out_wmma(const unsigned short* __restrict__ hnew,   // [N, S*L] bf16
                   const unsigned short* __restrict__ wy,     // [S, M, L] bf16
                   const float* __restrict__ by,              // [S, M]
                   float* __restrict__ y) {                   // [N, S*M]
  const int s      = blockIdx.z;
  const int m0     = blockIdx.x * 64;
  const int n0     = blockIdx.y * 128;
  const int wave   = threadIdx.x >> 5;
  const int lane   = threadIdx.x & 31;
  const int laneLo = lane & 15;
  const int hiHalf = lane >> 4;

  const int arow = n0 + wave * 16 + laneLo;
  const unsigned short* aptr = hnew + ((size_t)arow * SS + s) * LL;
  const int ak = hiHalf ? 16 : 0;
  const int bk = hiHalf ? 16 : 0;
  const unsigned short* bcol0 =
      wy + ((size_t)s * MM + m0 + laneLo) * LL + bk;   // + t*16*LL + k0 later

  v8f acc[4];
  const v8f vzero = {0.f, 0.f, 0.f, 0.f, 0.f, 0.f, 0.f, 0.f};
  #pragma unroll
  for (int t = 0; t < 4; ++t) acc[t] = vzero;

  for (int k0 = 0; k0 < LL; k0 += 32) {
    FragBF16 a;
    a.h[0] = *(const v8bf*)(aptr + k0 + ak);
    a.h[1] = *(const v8bf*)(aptr + k0 + ak + 8);
    FragBF16 b0, b1;
    b0.h[0] = *(const v8bf*)(bcol0 + k0);
    b0.h[1] = *(const v8bf*)(bcol0 + k0 + 8);
    #pragma unroll
    for (int t = 0; t < 4; ++t) {
      FragBF16& bc = (t & 1) ? b1 : b0;
      FragBF16& bn = (t & 1) ? b0 : b1;
      if (t + 1 < 4) {
        const unsigned short* bp = bcol0 + (size_t)(t + 1) * 16 * LL + k0;
        bn.h[0] = *(const v8bf*)(bp);
        bn.h[1] = *(const v8bf*)(bp + 8);
      }
      acc[t] = __builtin_amdgcn_wmma_f32_16x16x32_bf16(
          false, a.f, false, bc.f, (short)0, acc[t], false, false);
    }
  }

  #pragma unroll
  for (int t = 0; t < 4; ++t) {
    const int col = m0 + t * 16 + laneLo;
    const float vb = by[s * MM + col];
    #pragma unroll
    for (int j = 0; j < 8; ++j) {
      const int row = n0 + wave * 16 + j + (hiHalf ? 8 : 0);
      y[(size_t)row * (SS * MM) + s * MM + col] = acc[t][j] + vb;
    }
  }
}

// ---------------------------------------------------------------------------
// Host launcher
// ---------------------------------------------------------------------------
extern "C" void kernel_launch(void* const* d_in, const int* in_sizes, int n_in,
                              void* d_out, int out_size, void* d_ws, size_t ws_size,
                              hipStream_t stream) {
  (void)in_sizes; (void)n_in; (void)out_size; (void)ws_size;
  const float* modulation = (const float*)d_in[0];
  const float* h0  = (const float*)d_in[1];
  const float* c0  = (const float*)d_in[2];
  const float* Wi  = (const float*)d_in[3];
  const float* bi  = (const float*)d_in[4];
  const float* Wf  = (const float*)d_in[5];
  const float* bf  = (const float*)d_in[6];
  const float* Wg  = (const float*)d_in[7];
  const float* bg  = (const float*)d_in[8];
  const float* Wo  = (const float*)d_in[9];
  const float* bo  = (const float*)d_in[10];
  const float* Wy  = (const float*)d_in[11];
  const float* by  = (const float*)d_in[12];
  float* y = (float*)d_out;

  // Workspace partition (bf16 ushorts), total ~106 MB:
  unsigned short* xh = (unsigned short*)d_ws;            // N*S*D
  unsigned short* w4 = xh + (size_t)NN * SS * DD;        // 4*S*L*D
  unsigned short* wy = w4 + (size_t)4 * SS * LL * DD;    // S*M*L
  unsigned short* hn = wy + (size_t)SS * MM * LL;        // N*S*L

  const int SLD = SS * LL * DD;   // per-gate weight element count
  const int SML = SS * MM * LL;
  const int cvB = 256;

  build_xh_bf16<<<dim3(1, NN * SS), DD / 4, 0, stream>>>(modulation, h0, xh);
  conv_bf16x4<<<(SLD / 4 + cvB - 1) / cvB, cvB, 0, stream>>>(Wi, w4 + 0 * (size_t)SLD, SLD);
  conv_bf16x4<<<(SLD / 4 + cvB - 1) / cvB, cvB, 0, stream>>>(Wf, w4 + 1 * (size_t)SLD, SLD);
  conv_bf16x4<<<(SLD / 4 + cvB - 1) / cvB, cvB, 0, stream>>>(Wg, w4 + 2 * (size_t)SLD, SLD);
  conv_bf16x4<<<(SLD / 4 + cvB - 1) / cvB, cvB, 0, stream>>>(Wo, w4 + 3 * (size_t)SLD, SLD);
  conv_bf16x4<<<(SML / 4 + cvB - 1) / cvB, cvB, 0, stream>>>(Wy, wy, SML);

  lstm_gates_wmma<<<dim3(LL / 64, NN / 128, SS), 256, 0, stream>>>(
      xh, w4, bi, bf, bg, bo, c0, hn);
  lstm_out_wmma<<<dim3(MM / 64, NN / 128, SS), 256, 0, stream>>>(hn, wy, by, y);
}
